// NeuralCDE_12756052869411
// MI455X (gfx1250) — matrix operand order repaired and checked by reading, not compiled, not run on hardware
//
#include <hip/hip_runtime.h>
#include <hip/hip_bf16.h>

typedef __attribute__((ext_vector_type(16))) _Float16 v16h;
typedef __attribute__((ext_vector_type(8)))  _Float16 v8h;
typedef __attribute__((ext_vector_type(8)))  float    v8f;

#define BB    512
#define SS    1024
#define CIN   31
#define HH    64
#define OUTD  10
#define HC    2048   // H * C = 64 * 32
#define TPB   512    // 16 waves; wave w owns N-chunk [128w, 128w+128)

// Branch-free tanh: gfx1250 hardware v_tanh_f32.
__device__ __forceinline__ float fast_tanh(float x) {
#if __has_builtin(__builtin_amdgcn_tanhf)
  return __builtin_amdgcn_tanhf(x);
#elif __has_builtin(__builtin_amdgcn_tanh_f32)
  return __builtin_amdgcn_tanh_f32(x);
#else
  const float a = __builtin_fabsf(x);
  const float e = __builtin_amdgcn_exp2f(-2.8853900817779268f * a);
  const float t = (1.f - e) * __builtin_amdgcn_rcpf(1.f + e);
  return __builtin_copysignf(t, x);
#endif
}

// v + (v cross-lane via DPP quad_perm); ctrl must be a literal.
#define DPP_ADD(v, ctrl)                                                     \
  ((v) + __int_as_float(__builtin_amdgcn_update_dpp(                         \
             0, __float_as_int(v), (ctrl), 0xF, 0xF, true)))
#define QP_XOR1 0xB1  // quad_perm(1,0,3,2)
#define QP_XOR2 0x4E  // quad_perm(2,3,0,1)

// g(z) for a 16-batch tile. F_pre = z @ M^T (M = W2@W1 f16, B register-
// resident), F = tanh(F_pre), k[b][h] = sum_c F[b,h*32+c]*slope[b][c].
// Wave w owns h in [4w, 4w+4) -> 8 tiles of N=16.
__device__ __forceinline__ void eval_g(const _Float16* zh, float* kdst,
                                       const v16h (&Bf)[8][2],
                                       const float (&slp)[2][8],
                                       int wave, int g, int m) {
  // A fragments (contiguous k order, matching B): 16 contiguous halves per lane half per K-step.
  const v8h a0 = *(const v8h*)(zh + m * HH + (g << 4));
  const v8h a1 = *(const v8h*)(zh + m * HH + 8 + (g << 4));
  const v8h a2 = *(const v8h*)(zh + m * HH + 32 + (g << 4));
  const v8h a3 = *(const v8h*)(zh + m * HH + 40 + (g << 4));
  const v16h Alo = __builtin_shufflevector(a0, a1, 0, 1, 2, 3, 4, 5, 6, 7,
                                           8, 9, 10, 11, 12, 13, 14, 15);
  const v16h Ahi = __builtin_shufflevector(a2, a3, 0, 1, 2, 3, 4, 5, 6, 7,
                                           8, 9, 10, 11, 12, 13, 14, 15);
  const int hbase = wave << 2;
#pragma unroll
  for (int u = 0; u < 4; ++u) {          // 4 h-values per wave
    float hp[8] = {0.f, 0.f, 0.f, 0.f, 0.f, 0.f, 0.f, 0.f};
#pragma unroll
    for (int ch = 0; ch < 2; ++ch) {     // two 16-wide c halves per h
      v8f acc = {};
      acc = __builtin_amdgcn_wmma_f32_16x16x32_f16(
          false, Alo, false, Bf[2 * u + ch][0], (short)0, acc, false, false);
      acc = __builtin_amdgcn_wmma_f32_16x16x32_f16(
          false, Ahi, false, Bf[2 * u + ch][1], (short)0, acc, false, false);
      // D layout: VGPR r -> row (r + 8*g), col = lane%16
#pragma unroll
      for (int r = 0; r < 8; ++r) hp[r] += fast_tanh(acc[r]) * slp[ch][r];
    }
    // Packed multi-reduce over 16 c-lanes; final S[r] lands in lane m==r.
    float p[4], q[2];
#pragma unroll
    for (int j = 0; j < 4; ++j) {        // level 0: xor1 (DPP quad_perm)
      const float a = DPP_ADD(hp[2 * j], QP_XOR1);
      const float b = DPP_ADD(hp[2 * j + 1], QP_XOR1);
      p[j] = (m & 1) ? b : a;
    }
#pragma unroll
    for (int j = 0; j < 2; ++j) {        // level 1: xor2 (DPP quad_perm)
      const float a = DPP_ADD(p[2 * j], QP_XOR2);
      const float b = DPP_ADD(p[2 * j + 1], QP_XOR2);
      q[j] = (m & 2) ? b : a;
    }
    // levels 2-3: xor4 / xor8 (issue both chains before consuming)
    const float t0 = __shfl_xor(q[0], 4);
    const float t1 = __shfl_xor(q[1], 4);
    q[0] += t0;
    q[1] += t1;
    const float r0 = q[0] + __shfl_xor(q[0], 8);
    const float r1 = q[1] + __shfl_xor(q[1], 8);
    const float fin = (m & 4) ? r1 : r0;  // lane m holds S[m&7]
    if (m < 8) kdst[(m + (g << 3)) * HH + hbase + u] = fin;
  }
}

// M = W2 @ W1 : [2048,4] x [4,64] -> [2048,64], stored f16 row-major.
__global__ void cde_prep(const float* __restrict__ W1,
                         const float* __restrict__ W2,
                         _Float16* __restrict__ Mt) {
  const int i = blockIdx.x * 256 + threadIdx.x;
  if (i >= HC * HH) return;
  const int n = i >> 6, k = i & 63;
  float a = 0.f;
#pragma unroll
  for (int j = 0; j < 4; ++j) a += W2[n * 4 + j] * W1[j * HH + k];
  Mt[i] = (_Float16)a;
}

__global__ __launch_bounds__(TPB) void cde_main(
    const float* __restrict__ inp, const float* __restrict__ z0,
    const float* __restrict__ Wd, const float* __restrict__ bd,
    const _Float16* __restrict__ Mt, float* __restrict__ out) {
  __shared__ float sZ[16 * HH];                     // f32 state
  __shared__ __align__(16) _Float16 sZh[16 * HH];   // f16 stage input (A operand)
  __shared__ float sK[4][16 * HH];
  __shared__ float sSlope[16 * 32];
  __shared__ float sX[2][16 * 32];
  __shared__ float sWd[OUTD * HH];
  __shared__ float sBd[OUTD];

  const int tid = threadIdx.x;
  const int wave = tid >> 5, lane = tid & 31;
  const int g = lane >> 4, m = lane & 15;
  const int bT = blockIdx.x * 16;
  const float dt = 2.0f / 1023.0f;
  const float invdt = 1023.0f / 2.0f;

  // Preload this wave's 16 B fragments (8 tiles x 2 K-halves): 128 VGPRs,
  // below the 256-VGPR direct window (no MSB paging). Per lane a contiguous
  // 16-half slice of an M row; resident across all 1023 steps.
  v16h Bf[8][2];
#pragma unroll
  for (int t = 0; t < 8; ++t) {
    const size_t n = (size_t)((wave << 7) + (t << 4) + m);
    Bf[t][0] = *(const v16h*)(Mt + n * HH + (g << 4));
    Bf[t][1] = *(const v16h*)(Mt + n * HH + 32 + (g << 4));
  }

  for (int i = tid; i < 16 * HH; i += TPB) {
    const float v = z0[(size_t)(bT + (i >> 6)) * HH + (i & 63)];
    sZ[i] = v;
    sZh[i] = (_Float16)v;
  }
  for (int i = tid; i < OUTD * HH; i += TPB) sWd[i] = Wd[i];
  if (tid < OUTD) sBd[tid] = bd[tid];
  for (int i = tid; i < 16 * CIN; i += TPB) {
    const int b = i / CIN, c = i % CIN;
    sX[0][b * 32 + c + 1] = inp[(size_t)(bT + b) * SS * CIN + c];
  }
  __syncthreads();

  // Head for s = 0 (traj includes z0).
  for (int p = tid; p < 16 * OUTD; p += TPB) {
    const int b = p / OUTD, o = p % OUTD;
    float a = sBd[o];
#pragma unroll
    for (int h = 0; h < HH; ++h) a += sZ[b * HH + h] * sWd[o * HH + h];
    out[((size_t)(bT + b) * SS) * OUTD + o] = a;
  }

  int cur = 0;
#pragma unroll 1
  for (int s = 0; s < SS - 1; ++s) {
    const int nxt = cur ^ 1;
    for (int i = tid; i < 16 * CIN; i += TPB) {
      const int b = i / CIN, c = i % CIN;
      sX[nxt][b * 32 + c + 1] = inp[((size_t)(bT + b) * SS + s + 1) * CIN + c];
    }
    if (s + 2 < SS && tid < 16)  // next step's rows -> global_prefetch_b8
      __builtin_prefetch(inp + ((size_t)(bT + tid) * SS + s + 2) * CIN, 0, 0);
    __syncthreads();
    for (int i = tid; i < 16 * 32; i += TPB) {
      const int c = i & 31;
      sSlope[i] = (c == 0) ? 1.0f : (sX[nxt][i] - sX[cur][i]) * invdt;
    }
    __syncthreads();

    // Per-lane einsum slopes, constant across the 4 RK stages of this step.
    float slp[2][8];
#pragma unroll
    for (int ch = 0; ch < 2; ++ch)
#pragma unroll
      for (int r = 0; r < 8; ++r)
        slp[ch][r] = sSlope[(r + (g << 3)) * 32 + (ch << 4) + m];

    // RK4 (3/8 rule)
    eval_g(sZh, sK[0], Bf, slp, wave, g, m);
    __syncthreads();
    for (int i = tid; i < 16 * HH; i += TPB)
      sZh[i] = (_Float16)(sZ[i] + (dt / 3.f) * sK[0][i]);
    __syncthreads();
    eval_g(sZh, sK[1], Bf, slp, wave, g, m);
    __syncthreads();
    for (int i = tid; i < 16 * HH; i += TPB)
      sZh[i] = (_Float16)(sZ[i] + dt * sK[1][i] - (dt / 3.f) * sK[0][i]);
    __syncthreads();
    eval_g(sZh, sK[2], Bf, slp, wave, g, m);
    __syncthreads();
    for (int i = tid; i < 16 * HH; i += TPB)
      sZh[i] = (_Float16)(sZ[i] + dt * (sK[0][i] - sK[1][i] + sK[2][i]));
    __syncthreads();
    eval_g(sZh, sK[3], Bf, slp, wave, g, m);
    __syncthreads();
    for (int i = tid; i < 16 * HH; i += TPB) {
      const float v =
          sZ[i] + dt * 0.125f * (sK[0][i] + 3.f * (sK[1][i] + sK[2][i]) + sK[3][i]);
      sZ[i] = v;
      sZh[i] = (_Float16)v;
    }
    __syncthreads();

    // Dense head for timestep s+1, straight from LDS state.
    for (int p = tid; p < 16 * OUTD; p += TPB) {
      const int b = p / OUTD, o = p % OUTD;
      float a = sBd[o];
#pragma unroll
      for (int h = 0; h < HH; ++h) a += sZ[b * HH + h] * sWd[o * HH + h];
      out[((size_t)(bT + b) * SS + s + 1) * OUTD + o] = a;
    }
    cur = nxt;
  }
}

extern "C" void kernel_launch(void* const* d_in, const int* in_sizes, int n_in,
                              void* d_out, int out_size, void* d_ws, size_t ws_size,
                              hipStream_t stream) {
  const float* inp = (const float*)d_in[0];  // [512,1024,31]
  const float* z0  = (const float*)d_in[1];  // [512,64]
  const float* W1  = (const float*)d_in[2];  // [4,64]
  const float* W2  = (const float*)d_in[3];  // [2048,4]
  const float* Wd  = (const float*)d_in[4];  // [10,64]
  const float* bd  = (const float*)d_in[5];  // [10]
  _Float16* Mt = (_Float16*)d_ws;            // [2048,64] f16, 256 KB

  cde_prep<<<(HC * HH + 255) / 256, 256, 0, stream>>>(W1, W2, Mt);
  cde_main<<<BB / 16, TPB, 0, stream>>>(inp, z0, Wd, bd, Mt, (float*)d_out);
}
